// LocationSensitiveAttention_35862976922267
// MI455X (gfx1250) — compile-verified, matrix operand order checked by reading, not run on hardware
//
#include <hip/hip_runtime.h>
#include <hip/hip_bf16.h>

#define BB 32
#define TT 4096
#define EE 512
#define QQ 1024
#define AA 128
#define FF 32
#define KW 31

typedef __attribute__((ext_vector_type(16))) __bf16 v16bf;
typedef __attribute__((ext_vector_type(8)))  float  v8f;

union AFrag { unsigned int u[8]; uint4 q[2]; v16bf v; };

__device__ __forceinline__ unsigned int pack_bf16(float a, float b) {
    unsigned int ua = __float_as_uint(a);
    unsigned int ub = __float_as_uint(b);
    ua = (ua + 0x7FFFu + ((ua >> 16) & 1u)) >> 16;
    ub = (ub + 0x7FFFu + ((ub >> 16) & 1u)) >> 16;
    return (ub << 16) | (ua & 0xFFFFu);
}

// ---------------------------------------------------------------------------
// Pre-pack W_memory [512,128] fp32 into bf16 B-fragments:
// layout [kstep(16)][ntile(8)][lane(32)][pair(8)] (uint = 2 bf16, K even/odd)
// Per ISA 16-bit operand layout: lane<16 -> K base {0..7,16..23}, lane>=16 -> +8
// ---------------------------------------------------------------------------
__global__ __launch_bounds__(256) void pack_wmem_kernel(const float* __restrict__ Wm,
                                                        unsigned int* __restrict__ wfrag) {
    int idx  = blockIdx.x * 256 + threadIdx.x;      // 32768 total
    int p    = idx & 7;
    int lane = (idx >> 3) & 31;
    int j    = (idx >> 8) & 7;
    int ks   = idx >> 11;
    int n  = 16 * j + (lane & 15);
    int kb = 32 * ks + 2 * (p & 3) + ((p & 4) ? 16 : 0) + ((lane & 16) ? 8 : 0);
    float w0 = Wm[(size_t)kb * AA + n];
    float w1 = Wm[(size_t)(kb + 1) * AA + n];
    wfrag[idx] = pack_bf16(w0, w1);
}

// Same packing for W_loc [32,128] (single k-chunk): 2048 uints
__global__ __launch_bounds__(256) void pack_wloc_kernel(const float* __restrict__ Wl,
                                                        unsigned int* __restrict__ wlfrag) {
    int idx  = blockIdx.x * 256 + threadIdx.x;      // 2048 total
    int p    = idx & 7;
    int lane = (idx >> 3) & 31;
    int j    = idx >> 8;
    int n  = 16 * j + (lane & 15);
    int kb = 2 * (p & 3) + ((p & 4) ? 16 : 0) + ((lane & 16) ? 8 : 0);
    float w0 = Wl[(size_t)kb * AA + n];
    float w1 = Wl[(size_t)(kb + 1) * AA + n];
    wlfrag[idx] = pack_bf16(w0, w1);
}

// pq[b][a] = query[b,:] @ W_query[:,a]   (32 blocks x 128 threads)
__global__ __launch_bounds__(128) void pq_kernel(const float* __restrict__ query,
                                                 const float* __restrict__ Wq,
                                                 float* __restrict__ pq) {
    int b = blockIdx.x, a = threadIdx.x;
    float acc = 0.f;
    for (int q = 0; q < QQ; ++q)
        acc += query[b * QQ + q] * Wq[(size_t)q * AA + a];
    pq[b * AA + a] = acc;
}

// Location conv: loc[b,t,f] = sum_w sum_c attn_cat[b,t+w-15,c]*ck[w,c,f]
__global__ __launch_bounds__(256) void loc_conv_kernel(const float* __restrict__ prev,
                                                       const float* __restrict__ cum,
                                                       const float* __restrict__ ck,
                                                       float* __restrict__ locf) {
    __shared__ float prevS[128 + KW - 1];
    __shared__ float cumS[128 + KW - 1];
    __shared__ float ckS[KW * 2 * FF];
    int b = blockIdx.y, t0 = blockIdx.x * 128, tid = threadIdx.x;
    for (int i = tid; i < 128 + KW - 1; i += 256) {
        int t = t0 + i - 15;
        float pv = 0.f, cv = 0.f;
        if (t >= 0 && t < TT) { pv = prev[b * TT + t]; cv = cum[b * TT + t]; }
        prevS[i] = pv; cumS[i] = cv;
    }
    for (int i = tid; i < KW * 2 * FF; i += 256) ckS[i] = ck[i];
    __syncthreads();
    for (int i = 0; i < 16; ++i) {
        int item = tid + 256 * i;
        int tl = item >> 5, f = item & 31;
        float acc = 0.f;
        #pragma unroll
        for (int w = 0; w < KW; ++w)
            acc += prevS[tl + w] * ckS[(w * 2 + 0) * FF + f]
                 + cumS[tl + w]  * ckS[(w * 2 + 1) * FF + f];
        locf[((size_t)b * TT + t0 + tl) * FF + f] = acc;
    }
}

// ---------------------------------------------------------------------------
// Main fused kernel: energies[b,t] = tanh( mem@Wm + loc@Wl + pq ) @ Wv, masked.
// Block = (t-tile of 128, batch b), 256 threads = 8 waves; each wave owns a
// 16-row strip and 8 accumulator tiles (full N=128). K = 16 chunks of mem +
// 1 chunk of loc, all via v_wmma_f32_16x16x32_bf16.
// ---------------------------------------------------------------------------
__global__ __launch_bounds__(256) void energies_kernel(const float* __restrict__ mem,
                                                       const unsigned char* __restrict__ mask,
                                                       const float* __restrict__ pq,
                                                       const unsigned int* __restrict__ wfrag,
                                                       const unsigned int* __restrict__ wlfrag,
                                                       const float* __restrict__ locf,
                                                       const float* __restrict__ Wv,
                                                       float* __restrict__ energ) {
    __shared__ unsigned int aTile[128 * 17];     // 128 rows x 16 k-pairs (+1 pad)
    __shared__ unsigned int bTile[8 * 32 * 8];   // prepacked B fragments

    int b = blockIdx.y, t0 = blockIdx.x * 128;
    int tid = threadIdx.x, lane = tid & 31, w = tid >> 5;

    v8f c[8] = {};

    for (int kc = 0; kc < 17; ++kc) {
        // ---- stage A tile (fp32 -> packed bf16) + B fragments ----
        if (kc < 16) {
            const float* src = mem + ((size_t)b * TT + t0) * EE + kc * 32;
            #pragma unroll
            for (int i = 0; i < 4; ++i) {
                int f4 = tid + 256 * i;          // 1024 float4 = 128 rows x 8
                int row = f4 >> 3, kk4 = f4 & 7;
                float4 v = *(const float4*)(src + (size_t)row * EE + kk4 * 4);
                float m = mask[b * TT + t0 + row] ? 1.f : 0.f;
                aTile[row * 17 + kk4 * 2]     = pack_bf16(v.x * m, v.y * m);
                aTile[row * 17 + kk4 * 2 + 1] = pack_bf16(v.z * m, v.w * m);
            }
            const uint4* wsrc = (const uint4*)(wfrag + kc * 2048);
            uint4* bdst = (uint4*)bTile;
            bdst[tid]       = wsrc[tid];
            bdst[tid + 256] = wsrc[tid + 256];
        } else {
            const float* src = locf + ((size_t)b * TT + t0) * FF;
            #pragma unroll
            for (int i = 0; i < 4; ++i) {
                int f4 = tid + 256 * i;
                int row = f4 >> 3, kk4 = f4 & 7;
                float4 v = *(const float4*)(src + (size_t)row * FF + kk4 * 4);
                aTile[row * 17 + kk4 * 2]     = pack_bf16(v.x, v.y);
                aTile[row * 17 + kk4 * 2 + 1] = pack_bf16(v.z, v.w);
            }
            const uint4* wsrc = (const uint4*)wlfrag;
            uint4* bdst = (uint4*)bTile;
            bdst[tid]       = wsrc[tid];
            bdst[tid + 256] = wsrc[tid + 256];
        }
        __syncthreads();

        // ---- A fragment for this wave's 16-row strip ----
        AFrag a;
        int row = w * 16 + (lane & 15);
        #pragma unroll
        for (int p = 0; p < 8; ++p)
            a.u[p] = aTile[row * 17 + (p & 3) + ((p & 4) << 1) + ((lane & 16) >> 2)];

        // ---- 8 WMMAs (full N=128) ----
        #pragma unroll
        for (int j = 0; j < 8; ++j) {
            AFrag bf;
            const uint4* bp = (const uint4*)&bTile[(j * 32 + lane) * 8];
            bf.q[0] = bp[0];
            bf.q[1] = bp[1];
            c[j] = __builtin_amdgcn_wmma_f32_16x16x32_bf16(
                false, a.v, false, bf.v, (short)0, c[j], false, false);
        }
        __syncthreads();
    }

    // ---- epilogue: +pq, tanh, dot with Wv, reduce across N ----
    float partial[8] = {0.f, 0.f, 0.f, 0.f, 0.f, 0.f, 0.f, 0.f};
    #pragma unroll
    for (int j = 0; j < 8; ++j) {
        int n = 16 * j + (lane & 15);
        float pqn = pq[b * AA + n];
        float wvn = Wv[n];
        #pragma unroll
        for (int r = 0; r < 8; ++r)
            partial[r] += tanhf(c[j][r] + pqn) * wvn;
    }
    // sum across 16 lanes of each half (same C-matrix M, different N groups)
    #pragma unroll
    for (int r = 0; r < 8; ++r) {
        #pragma unroll
        for (int off = 1; off < 16; off <<= 1)
            partial[r] += __shfl_xor(partial[r], off, 32);
    }
    if ((lane & 15) == 0) {
        int rowBase = t0 + w * 16 + ((lane & 16) ? 8 : 0);  // C: lanes>=16 -> M+8
        #pragma unroll
        for (int r = 0; r < 8; ++r) {
            int t = rowBase + r;
            float e = partial[r];
            energ[b * TT + t] = mask[b * TT + t] ? e : -3.402823466e38f;
        }
    }
}

// Row softmax over T, plus cumulative update. One block per batch row.
__global__ __launch_bounds__(256) void softmax_kernel(const float* __restrict__ energ,
                                                      const float* __restrict__ prev_cum,
                                                      float* __restrict__ attn,
                                                      float* __restrict__ cum) {
    __shared__ float red[256];
    __shared__ float sval;
    int b = blockIdx.x, tid = threadIdx.x;
    float m = -3.402823466e38f;
    for (int t = tid; t < TT; t += 256) m = fmaxf(m, energ[b * TT + t]);
    red[tid] = m; __syncthreads();
    for (int s = 128; s > 0; s >>= 1) {
        if (tid < s) red[tid] = fmaxf(red[tid], red[tid + s]);
        __syncthreads();
    }
    if (tid == 0) sval = red[0];
    __syncthreads();
    float mx = sval;
    __syncthreads();
    float ssum = 0.f;
    for (int t = tid; t < TT; t += 256) ssum += __expf(energ[b * TT + t] - mx);
    red[tid] = ssum; __syncthreads();
    for (int s = 128; s > 0; s >>= 1) {
        if (tid < s) red[tid] += red[tid + s];
        __syncthreads();
    }
    if (tid == 0) sval = red[0];
    __syncthreads();
    float inv = 1.f / sval;
    for (int t = tid; t < TT; t += 256) {
        float a = __expf(energ[b * TT + t] - mx) * inv;
        attn[b * TT + t] = a;
        cum[b * TT + t]  = a + prev_cum[b * TT + t];
    }
}

// context partials over 256-wide t chunks: ctxp[b][chunk][e]
__global__ __launch_bounds__(256) void context_part_kernel(const float* __restrict__ attn,
                                                           const float* __restrict__ mem,
                                                           const unsigned char* __restrict__ mask,
                                                           float* __restrict__ ctxp) {
    __shared__ float wS[256];
    int b = blockIdx.y, t0 = blockIdx.x * 256, tid = threadIdx.x;
    int t = t0 + tid;
    float a = attn[b * TT + t];
    wS[tid] = mask[b * TT + t] ? a : 0.f;
    __syncthreads();
    #pragma unroll
    for (int half = 0; half < 2; ++half) {
        int e = tid + 256 * half;
        float acc = 0.f;
        for (int ttl = 0; ttl < 256; ++ttl)
            acc += wS[ttl] * mem[((size_t)b * TT + t0 + ttl) * EE + e];
        ctxp[((size_t)b * 16 + blockIdx.x) * EE + e] = acc;
    }
}

__global__ __launch_bounds__(256) void context_reduce_kernel(const float* __restrict__ ctxp,
                                                             float* __restrict__ ctx) {
    int idx = blockIdx.x * 256 + threadIdx.x;   // B*E = 16384
    int b = idx >> 9, e = idx & 511;
    float acc = 0.f;
    #pragma unroll
    for (int ch = 0; ch < 16; ++ch)
        acc += ctxp[((size_t)b * 16 + ch) * EE + e];
    ctx[idx] = acc;
}

extern "C" void kernel_launch(void* const* d_in, const int* in_sizes, int n_in,
                              void* d_out, int out_size, void* d_ws, size_t ws_size,
                              hipStream_t stream) {
    const float* query    = (const float*)d_in[0];
    const float* memory   = (const float*)d_in[1];
    const unsigned char* mask = (const unsigned char*)d_in[2];
    const float* prev     = (const float*)d_in[3];
    const float* prev_cum = (const float*)d_in[4];
    const float* Wq       = (const float*)d_in[5];
    const float* Wm       = (const float*)d_in[6];
    const float* Wv       = (const float*)d_in[7];
    const float* ck       = (const float*)d_in[8];
    const float* Wl       = (const float*)d_in[9];

    // workspace layout
    float*        pq_ws  = (float*)d_ws;                          // 4096 f
    unsigned int* wfrag  = (unsigned int*)(pq_ws + 4096);         // 32768 u
    unsigned int* wlfrag = wfrag + 32768;                         // 2048 u
    float*        energ  = (float*)(wlfrag + 2048);               // B*T f
    float*        locf   = energ + (size_t)BB * TT;               // B*T*F f
    float*        ctxp   = locf + (size_t)BB * TT * FF;           // B*16*E f

    float* ctx_out  = (float*)d_out;                              // B*E
    float* attn_out = ctx_out + BB * EE;                          // B*T
    float* cum_out  = attn_out + BB * TT;                         // B*T

    pack_wmem_kernel<<<128, 256, 0, stream>>>(Wm, wfrag);
    pack_wloc_kernel<<<8, 256, 0, stream>>>(Wl, wlfrag);
    pq_kernel<<<BB, 128, 0, stream>>>(query, Wq, pq_ws);
    loc_conv_kernel<<<dim3(TT / 128, BB), 256, 0, stream>>>(prev, prev_cum, ck, locf);
    energies_kernel<<<dim3(TT / 128, BB), 256, 0, stream>>>(memory, mask, pq_ws, wfrag,
                                                            wlfrag, locf, Wv, energ);
    softmax_kernel<<<BB, 256, 0, stream>>>(energ, prev_cum, attn_out, cum_out);
    context_part_kernel<<<dim3(16, BB), 256, 0, stream>>>(attn_out, memory, mask, ctxp);
    context_reduce_kernel<<<64, 256, 0, stream>>>(ctxp, ctx_out);
}
